// SpatialVectorQuantizer0_8254927142942
// MI455X (gfx1250) — matrix operand order repaired and checked by reading, not compiled, not run on hardware
//
#include <hip/hip_runtime.h>

#define S_DIM 1024
#define D_DIM 4096
#define K_EMB 2048

typedef __attribute__((ext_vector_type(16))) __bf16 v16bf;
typedef __attribute__((ext_vector_type(8)))  __bf16 v8bf;
typedef __attribute__((ext_vector_type(2)))  __bf16 v2bf;
typedef __attribute__((ext_vector_type(8)))  float  v8f;
typedef __attribute__((ext_vector_type(2)))  float  v2f;

// pack two floats into two bf16 (one dword) -> single v_cvt_pk_bf16_f32
__device__ __forceinline__ unsigned pk2bf(float f0, float f1) {
  v2f v = {f0, f1};
  v2bf b = __builtin_convertvector(v, v2bf);
  return __builtin_bit_cast(unsigned, b);
}

// Build a v16bf fragment from two aligned 16B LDS chunks (2x ds_load_b128).
__device__ __forceinline__ v16bf frag16(const __bf16* p0, const __bf16* p1) {
  v8bf lo = *(const v8bf*)p0;
  v8bf hi = *(const v8bf*)p1;
  v16bf r;
#pragma unroll
  for (int i = 0; i < 8; ++i) { r[i] = lo[i]; r[i + 8] = hi[i]; }
  return r;
}

// ---------------- kernel 1: embedding row norms ----------------
__global__ __launch_bounds__(256) void vq_norms(const float* __restrict__ E,
                                                float* __restrict__ en) {
  __shared__ float red[256];
  const int k = blockIdx.x, t = threadIdx.x;
  const float4* row = (const float4*)(E + (size_t)k * D_DIM);
  float a = 0.f;
#pragma unroll
  for (int i = 0; i < 4; ++i) {
    float4 v = row[t + i * 256];
    a += v.x * v.x + v.y * v.y + v.z * v.z + v.w * v.w;
  }
  red[t] = a; __syncthreads();
  for (int s = 128; s > 0; s >>= 1) { if (t < s) red[t] += red[t + s]; __syncthreads(); }
  if (t == 0) en[k] = red[0];
}

// ---------------- kernel 2: WMMA distance GEMM + per-tile argmin ----------------
// dist[s,k] = en[k] - 2 * sum_d x[s,d]*E[k,d]   (||x||^2 dropped: constant per row)
// x[s,d] = in[d*S + s].  Double-buffered: regs hold tile kt+1's global data while
// WMMAs run on LDS tile kt; two LDS buffers -> one barrier per K-step.
__global__ __launch_bounds__(256) void vq_dist_wmma(const float* __restrict__ in,
                                                    const float* __restrict__ E,
                                                    const float* __restrict__ en,
                                                    float* __restrict__ cand_v,
                                                    int* __restrict__ cand_i) {
  __shared__ __bf16 As[2][64][48];   // 48-elem (96B) rows: aligned b128 chunks
  __shared__ __bf16 Bs[2][128][48];
  __shared__ float cvv[64][2];
  __shared__ int   cii[64][2];

  const int t = threadIdx.x;
  const int lane = t & 31, wave = t >> 5;
  const int wm = wave & 3, wn = wave >> 2;       // 4 s-groups x 2 k-halves
  const int h = lane >> 4, ln = lane & 15;
  const int s0 = blockIdx.x * 64;
  const int k0 = blockIdx.y * 128;

  const v8f vzero = {0.f, 0.f, 0.f, 0.f, 0.f, 0.f, 0.f, 0.f};
  v8f acc[4];
#pragma unroll
  for (int q = 0; q < 4; ++q) acc[q] = vzero;

  float  ra[8];   // staged A floats (4 dword-pairs)
  float2 rb[8];   // staged B float2 (8 dword-pairs)

  auto load_tile = [&](int kt) {
    const int d0 = kt * 32;
#pragma unroll
    for (int i = 0; i < 4; ++i) {          // A: 64 s x 32 d, s contiguous in mem
      int lin = i * 256 + t, sl = lin & 63, dp = lin >> 6;
      ra[2 * i]     = in[(size_t)(d0 + 2 * dp)     * S_DIM + s0 + sl];
      ra[2 * i + 1] = in[(size_t)(d0 + 2 * dp + 1) * S_DIM + s0 + sl];
    }
#pragma unroll
    for (int i = 0; i < 8; ++i) {          // B: 128 k x 32 d, d contiguous -> b64
      int lin = i * 256 + t, dp = lin & 15, nl = lin >> 4;
      rb[i] = *(const float2*)&E[(size_t)(k0 + nl) * D_DIM + d0 + 2 * dp];
    }
  };
  auto store_tile = [&](int b) {
#pragma unroll
    for (int i = 0; i < 4; ++i) {
      int lin = i * 256 + t, sl = lin & 63, dp = lin >> 6;
      *(unsigned*)&As[b][sl][2 * dp] = pk2bf(ra[2 * i], ra[2 * i + 1]);
    }
#pragma unroll
    for (int i = 0; i < 8; ++i) {
      int lin = i * 256 + t, dp = lin & 15, nl = lin >> 4;
      *(unsigned*)&Bs[b][nl][2 * dp] = pk2bf(rb[i].x, rb[i].y);
    }
  };
  auto compute = [&](int b) {
    // A fragment: lane holds row m, kappa in {8h..8h+7} U {16+8h..16+8h+7}
    const int m = wm * 16 + ln;
    v16bf a = frag16(&As[b][m][8 * h], &As[b][m][16 + 8 * h]);
    v16bf bf[4];
#pragma unroll
    for (int q = 0; q < 4; ++q) {          // preload all B frags -> staggered waits
      const int n = wn * 64 + q * 16 + ln;
      bf[q] = frag16(&Bs[b][n][8 * h], &Bs[b][n][16 + 8 * h]);
    }
#pragma unroll
    for (int q = 0; q < 4; ++q)
      acc[q] = __builtin_amdgcn_wmma_f32_16x16x32_bf16(
          false, a, false, bf[q], (short)0, acc[q], false, false);
  };

  load_tile(0);
  for (int kt = 0; kt < D_DIM / 32; kt += 2) {
    store_tile(0);                         // consumes loads issued last iteration
    load_tile(kt + 1);                     // in flight during compute(0)
    __syncthreads();
    compute(0);
    store_tile(1);
    if (kt + 2 < D_DIM / 32) load_tile(kt + 2);   // in flight during compute(1)
    __syncthreads();
    compute(1);
  }

  // Epilogue: dist = en[k] - 2*acc; local argmin over this block's 128 k per s.
  // C layout: vgpr r at lane -> row m = 8h + r, col n = ln.
  float bestv[8]; int besti[8];
#pragma unroll
  for (int q = 0; q < 4; ++q) {
    const int kg = k0 + wn * 64 + q * 16 + ln;
    const float e2 = en[kg];
#pragma unroll
    for (int r = 0; r < 8; ++r) {
      float v = e2 - 2.0f * acc[q][r];
      if (q == 0) { bestv[r] = v; besti[r] = kg; }
      else if (v < bestv[r]) { bestv[r] = v; besti[r] = kg; }  // ties keep lower k
    }
  }
  // reduce across the 16 lanes sharing the same s (xor masks stay within half)
#pragma unroll
  for (int r = 0; r < 8; ++r) {
#pragma unroll
    for (int off = 1; off < 16; off <<= 1) {
      float ov = __shfl_xor(bestv[r], off, 32);
      int   oi = __shfl_xor(besti[r], off, 32);
      if (ov < bestv[r] || (ov == bestv[r] && oi < besti[r])) { bestv[r] = ov; besti[r] = oi; }
    }
  }
  if (ln == 0) {
#pragma unroll
    for (int r = 0; r < 8; ++r) {
      const int sl = wm * 16 + 8 * h + r;
      cvv[sl][wn] = bestv[r];
      cii[sl][wn] = besti[r];
    }
  }
  __syncthreads();
  if (t < 64) {
    float v0 = cvv[t][0], v1 = cvv[t][1];
    int   i0 = cii[t][0], i1 = cii[t][1];
    bool take1 = (v1 < v0);   // wn=1 indices are larger, so ties keep wn=0
    cand_v[(size_t)(s0 + t) * 16 + blockIdx.y] = take1 ? v1 : v0;
    cand_i[(size_t)(s0 + t) * 16 + blockIdx.y] = take1 ? i1 : i0;
  }
}

// ---------------- kernel 3: final argmin across 16 k-tiles ----------------
__global__ __launch_bounds__(256) void vq_argmin(const float* __restrict__ cand_v,
                                                 const int* __restrict__ cand_i,
                                                 int* __restrict__ idx_i,
                                                 float* __restrict__ idx_f) {
  const int s = blockIdx.x * 256 + threadIdx.x;
  if (s >= S_DIM) return;
  const float* cv = cand_v + (size_t)s * 16;
  const int*   ci = cand_i + (size_t)s * 16;
  float bv = cv[0]; int bi = ci[0];
#pragma unroll
  for (int j = 1; j < 16; ++j) {
    float v = cv[j]; int ii = ci[j];
    if (v < bv || (v == bv && ii < bi)) { bv = v; bi = ii; }
  }
  idx_i[s] = bi;
  idx_f[s] = (float)bi;
}

// ---------------- kernel 4: gather q = E[idx], write output, partial loss ----------------
// out[b,l,s] flat index == d*S + s  (same layout as in), q_ste = x + (q - x)
__global__ __launch_bounds__(256) void vq_gather(const float* __restrict__ in,
                                                 const float* __restrict__ E,
                                                 const int* __restrict__ idx_i,
                                                 float* __restrict__ outq,
                                                 float* __restrict__ partial) {
  __shared__ float Et[32][33];
  __shared__ int   sidx[32];
  __shared__ float red[256];
  const int t = threadIdx.x;
  const int s0 = blockIdx.x * 32;
  const int d0 = blockIdx.y * 32;
  if (t < 32) sidx[t] = idx_i[s0 + t];
  __syncthreads();
  // coalesced E gather (d contiguous per row)
#pragma unroll
  for (int i = 0; i < 4; ++i) {
    int lin = i * 256 + t;
    int dl = lin & 31, sl = lin >> 5;
    Et[sl][dl] = E[(size_t)sidx[sl] * D_DIM + d0 + dl];
  }
  __syncthreads();
  // coalesced in/out access (s contiguous)
  float a = 0.f;
#pragma unroll
  for (int i = 0; i < 4; ++i) {
    int lin = i * 256 + t;
    int sl = lin & 31, dl = lin >> 5;
    size_t g = (size_t)(d0 + dl) * S_DIM + s0 + sl;
    float q = Et[sl][dl];
    float x = in[g];
    float df = q - x;
    outq[g] = x + df;          // straight-through estimator, matches ref rounding
    a += df * df;              // full fp32 loss term
  }
  red[t] = a; __syncthreads();
  for (int s = 128; s > 0; s >>= 1) { if (t < s) red[t] += red[t + s]; __syncthreads(); }
  if (t == 0) partial[blockIdx.y * gridDim.x + blockIdx.x] = red[0];
}

// ---------------- kernel 5: loss + perplexity ----------------
__global__ __launch_bounds__(256) void vq_finalize(const float* __restrict__ partial,
                                                   float* __restrict__ d_out) {
  __shared__ float red[256];
  const int t = threadIdx.x;
  float a = 0.f;
#pragma unroll
  for (int i = 0; i < 16; ++i) a += partial[t + i * 256];
  red[t] = a; __syncthreads();
  for (int s = 128; s > 0; s >>= 1) { if (t < s) red[t] += red[t + s]; __syncthreads(); }
  if (t == 0) {
    // total_loss = (1 + 0.25) * sum((q-x)^2) / (S*D)
    d_out[0] = 1.25f * red[0] / ((float)S_DIM * (float)D_DIM);
    // perplexity: one-hot -> exp(-log(1+1e-10)) == 1.0f exactly in fp32
    d_out[1 + (size_t)S_DIM * D_DIM] = 1.0f;
  }
}

extern "C" void kernel_launch(void* const* d_in, const int* in_sizes, int n_in,
                              void* d_out, int out_size, void* d_ws, size_t ws_size,
                              hipStream_t stream) {
  (void)in_sizes; (void)n_in; (void)out_size; (void)ws_size;
  const float* in = (const float*)d_in[0];   // [8,512,1024] f32
  const float* E  = (const float*)d_in[1];   // [2048,4096] f32
  float* out = (float*)d_out;

  // workspace layout (~156KB total)
  float* en      = (float*)d_ws;                  // 2048 f32
  float* cand_v  = en + K_EMB;                    // 1024*16 f32
  int*   cand_i  = (int*)(cand_v + S_DIM * 16);   // 1024*16 i32
  int*   idx_i   = cand_i + S_DIM * 16;           // 1024 i32
  float* partial = (float*)(idx_i + S_DIM);       // 4096 f32

  float* outq  = out + 1;                          // quantized [B,L,S]
  float* idx_f = out + 2 + (size_t)S_DIM * D_DIM;  // idx after perplexity slot

  vq_norms   <<<K_EMB, 256, 0, stream>>>(E, en);
  vq_dist_wmma<<<dim3(S_DIM / 64, K_EMB / 128), 256, 0, stream>>>(in, E, en, cand_v, cand_i);
  vq_argmin  <<<S_DIM / 256, 256, 0, stream>>>(cand_v, cand_i, idx_i, idx_f);
  vq_gather  <<<dim3(S_DIM / 32, D_DIM / 32), 256, 0, stream>>>(in, E, idx_i, outq, partial);
  vq_finalize<<<1, 256, 0, stream>>>(partial, out);
}
